// Attention_13846974562678
// MI455X (gfx1250) — compile-verified, matrix operand order checked by reading, not run on hardware
//
#include <hip/hip_runtime.h>
#include <hip/hip_bf16.h>

// Problem constants (fixed by the reference)
#define BATCH 8
#define SEQ 1024
#define DIM 448
#define HEADS 7
#define DIM_HEAD 64
#define INNER 448          // HEADS * DIM_HEAD
#define QV_N 896           // 2 * INNER
#define ROWS (BATCH * SEQ) // 8192
#define SCALE_ATT 0.125f   // 64^-0.5

typedef __attribute__((ext_vector_type(16))) __bf16 v16bf;
typedef __attribute__((ext_vector_type(8)))  __bf16 v8bf;
typedef __attribute__((ext_vector_type(8)))  float  v8f;

// ---------------------------------------------------------------------------
// Fragment loader for 16-bit 16x32 A-matrix (and, by symmetry, Kx16 B-matrix
// loaded from a transposed [N][K] array): lane r = lane&15 picks the row
// (or B column), lane-half picks which K-half; elements 0..7 are K = h*8+e,
// elements 8..15 are K = 16 + h*8 + e  (per CDNA5 ISA 16-bit A
// layout). Two contiguous 16-byte loads per lane.
// ---------------------------------------------------------------------------
static __device__ __forceinline__ v16bf load_frag(const __bf16* base, int stride) {
  int lane = threadIdx.x & 31;
  int r = lane & 15, hh = lane >> 4;
  const __bf16* p = base + r * stride + hh * 8;
  v8bf lo = *(const v8bf*)(p);
  v8bf hi = *(const v8bf*)(p + 16);
  v16bf out;
#pragma unroll
  for (int i = 0; i < 8; ++i) { out[i] = lo[i]; out[8 + i] = hi[i]; }
  return out;
}

static __device__ __forceinline__ v8f wmma_bf16(v16bf a, v16bf b, v8f c) {
  // D = A*B + C, f32 accumulate, 16x16x32 bf16
  return __builtin_amdgcn_wmma_f32_16x16x32_bf16(false, a, false, b,
                                                 (short)0, c, false, false);
}

// ---------------------------------------------------------------------------
// fp32 -> bf16 converters (one-time, tiny vs. the bias stream)
// ---------------------------------------------------------------------------
__global__ void k_cvt_copy(const float* __restrict__ src, __bf16* __restrict__ dst, int n) {
  int i = blockIdx.x * blockDim.x + threadIdx.x;
  if (i < n) dst[i] = (__bf16)src[i];
}

__global__ void k_cvt_transpose(const float* __restrict__ src, __bf16* __restrict__ dst,
                                int rows, int cols) {
  int i = blockIdx.x * blockDim.x + threadIdx.x;
  if (i < rows * cols) {
    int r = i / cols, c = i - r * cols;
    dst[c * rows + r] = (__bf16)src[i]; // dst[N][K] = Bt
  }
}

// ---------------------------------------------------------------------------
// QV projection: [8192x448] @ [448x896] -> scattered into q[b,h,n,d] (bf16)
// and vT[b,h,d,n] (bf16). One 64x64 tile per wave (4x4 WMMA tiles):
// 8 fragment loads (16 x b128) feed 16 WMMAs per k-step -> 1 b128 per WMMA,
// ~205 MB total L2 read traffic for this GEMM (vs 410 MB at 32x32 tiling).
// ---------------------------------------------------------------------------
__global__ void __launch_bounds__(256)
k_qv_gemm(const __bf16* __restrict__ x, const __bf16* __restrict__ wqvT,
          __bf16* __restrict__ q_ws, __bf16* __restrict__ vT_ws) {
  int wave = blockIdx.x * (blockDim.x >> 5) + (threadIdx.x >> 5);
  int tm = wave & 127;        // ROWS/64 = 128 row tiles
  int tn = wave >> 7;         // QV_N/64 = 14 col tiles
  int row0 = tm * 64, col0 = tn * 64;

  v8f acc[4][4] = {};
#pragma unroll 1
  for (int k = 0; k < DIM; k += 32) {
    v16bf fa[4], fb[4];
#pragma unroll
    for (int mi = 0; mi < 4; ++mi)
      fa[mi] = load_frag(x + (size_t)(row0 + mi * 16) * DIM + k, DIM);
#pragma unroll
    for (int ni = 0; ni < 4; ++ni)
      fb[ni] = load_frag(wqvT + (size_t)(col0 + ni * 16) * DIM + k, DIM);
#pragma unroll
    for (int mi = 0; mi < 4; ++mi)
#pragma unroll
      for (int ni = 0; ni < 4; ++ni)
        acc[mi][ni] = wmma_bf16(fa[mi], fb[ni], acc[mi][ni]);
  }

  int lane = threadIdx.x & 31;
  int c = lane & 15, hh = lane >> 4;
#pragma unroll
  for (int mi = 0; mi < 4; ++mi)
#pragma unroll
    for (int ni = 0; ni < 4; ++ni)
#pragma unroll
      for (int j = 0; j < 8; ++j) {
        int rg = row0 + mi * 16 + j + 8 * hh;  // flat b*n row
        int cg = col0 + ni * 16 + c;           // column in [0,896)
        int bb = rg >> 10, n = rg & 1023;
        __bf16 bv = (__bf16)acc[mi][ni][j];
        if (cg < INNER) {                      // q: [b][h][n][d]
          int hd = cg >> 6, d = cg & 63;
          q_ws[(((size_t)(bb * HEADS + hd)) * SEQ + n) * DIM_HEAD + d] = bv;
        } else {                               // v: transposed [b][h][d][n]
          int cc = cg - INNER;
          int hd = cc >> 6, d = cc & 63;
          vT_ws[(((size_t)(bb * HEADS + hd)) * DIM_HEAD + d) * SEQ + n] = bv;
        }
      }
}

// ---------------------------------------------------------------------------
// Flash attention: one block per (b, h, 64-row q tile); 4 waves, each wave
// owns 16 q rows x 64 d. Online softmax over 32-key chunks; P re-layouted
// C->A through 1KB/wave of LDS. HBM-bound on the 235 MB fp32 bias stream.
// ---------------------------------------------------------------------------
__global__ void __launch_bounds__(128)
k_attn(const __bf16* __restrict__ q_ws, const __bf16* __restrict__ kx,
       const __bf16* __restrict__ vT_ws, const float* __restrict__ bias,
       __bf16* __restrict__ attn_out) {
  __shared__ __bf16 lds_p[4 * 16 * 32];

  int blk = blockIdx.x;        // b*HEADS*16 blocks
  int qt = blk & 15;
  int bh = blk >> 4;
  int head = bh % HEADS;
  int bb = bh / HEADS;
  int wave = threadIdx.x >> 5;
  int lane = threadIdx.x & 31;
  int c = lane & 15, hh = lane >> 4;
  int q0 = qt * 64 + wave * 16;

  const __bf16* qbase = q_ws + (((size_t)(bb * HEADS + head)) * SEQ + q0) * DIM_HEAD;
  const __bf16* kbase = kx + (size_t)head * SEQ * DIM_HEAD;
  const __bf16* vbase = vT_ws + ((size_t)(bb * HEADS + head)) * DIM_HEAD * SEQ;
  const float*  bbase = bias + ((size_t)head * SEQ + q0) * SEQ;

  v16bf qa0 = load_frag(qbase + 0, DIM_HEAD);   // q rows, d 0..31
  v16bf qa1 = load_frag(qbase + 32, DIM_HEAD);  // q rows, d 32..63

  v8f o[4] = {};
  float mrun[8], lrun[8];
#pragma unroll
  for (int j = 0; j < 8; ++j) { mrun[j] = -1e30f; lrun[j] = 0.f; }

  __bf16* myp = lds_p + wave * 16 * 32;

  for (int mk = 0; mk < SEQ; mk += 32) {
    // ---- S = q @ k_ext^T : two 16x16 key tiles, K(d)=64 via two k-steps ----
    v16bf kb0a = load_frag(kbase + (size_t)mk * DIM_HEAD + 0, DIM_HEAD);
    v16bf kb0b = load_frag(kbase + (size_t)mk * DIM_HEAD + 32, DIM_HEAD);
    v16bf kb1a = load_frag(kbase + (size_t)(mk + 16) * DIM_HEAD + 0, DIM_HEAD);
    v16bf kb1b = load_frag(kbase + (size_t)(mk + 16) * DIM_HEAD + 32, DIM_HEAD);
    v8f s0 = {}, s1 = {};
    s0 = wmma_bf16(qa0, kb0a, s0);
    s0 = wmma_bf16(qa1, kb0b, s0);
    s1 = wmma_bf16(qa0, kb1a, s1);
    s1 = wmma_bf16(qa1, kb1b, s1);

    // ---- bias (pre-scale, per reference) + scale; row-max across lanes ----
    float sv0[8], sv1[8], mx[8];
#pragma unroll
    for (int j = 0; j < 8; ++j) {
      int qr = j + 8 * hh;
      float bq0 = bbase[(size_t)qr * SEQ + mk + c];
      float bq1 = bbase[(size_t)qr * SEQ + mk + 16 + c];
      sv0[j] = (s0[j] + bq0) * SCALE_ATT;
      sv1[j] = (s1[j] + bq1) * SCALE_ATT;
      mx[j] = fmaxf(sv0[j], sv1[j]);
    }
#pragma unroll
    for (int m = 1; m < 16; m <<= 1)
#pragma unroll
      for (int j = 0; j < 8; ++j)
        mx[j] = fmaxf(mx[j], __shfl_xor(mx[j], m, 32));

    float alpha[8], psum[8];
#pragma unroll
    for (int j = 0; j < 8; ++j) {
      float mnew = fmaxf(mrun[j], mx[j]);
      alpha[j] = __expf(mrun[j] - mnew);
      mrun[j] = mnew;
      sv0[j] = __expf(sv0[j] - mnew);
      sv1[j] = __expf(sv1[j] - mnew);
      psum[j] = sv0[j] + sv1[j];
    }
#pragma unroll
    for (int m = 1; m < 16; m <<= 1)
#pragma unroll
      for (int j = 0; j < 8; ++j)
        psum[j] += __shfl_xor(psum[j], m, 32);
#pragma unroll
    for (int j = 0; j < 8; ++j)
      lrun[j] = lrun[j] * alpha[j] + psum[j];
#pragma unroll
    for (int dt = 0; dt < 4; ++dt)
#pragma unroll
      for (int j = 0; j < 8; ++j)
        o[dt][j] *= alpha[j];

    // ---- C-layout P -> LDS -> A-fragment (cross-lane, so wait DScnt) ----
#pragma unroll
    for (int j = 0; j < 8; ++j) {
      myp[(j + 8 * hh) * 32 + c]      = (__bf16)sv0[j];
      myp[(j + 8 * hh) * 32 + 16 + c] = (__bf16)sv1[j];
    }
    asm volatile("s_wait_dscnt 0" ::: "memory");
    v16bf pa = load_frag(myp, 32);

    // ---- O += P @ V  (B-fragments contiguous thanks to vT layout) ----
#pragma unroll
    for (int dt = 0; dt < 4; ++dt) {
      v16bf vb = load_frag(vbase + (size_t)(dt * 16) * SEQ + mk, SEQ);
      o[dt] = wmma_bf16(pa, vb, o[dt]);
    }
  }

  // ---- normalize and emit bf16 rows of [b*n][448] for the output GEMM ----
#pragma unroll
  for (int j = 0; j < 8; ++j) lrun[j] = 1.0f / lrun[j];
#pragma unroll
  for (int dt = 0; dt < 4; ++dt)
#pragma unroll
    for (int j = 0; j < 8; ++j) {
      int qr = q0 + j + 8 * hh;
      int cg = head * DIM_HEAD + dt * 16 + c;
      attn_out[((size_t)bb * SEQ + qr) * INNER + cg] = (__bf16)(o[dt][j] * lrun[j]);
    }
}

// ---------------------------------------------------------------------------
// Output projection: [8192x448] @ [448x448] + b_out -> fp32 d_out.
// 64x64 per wave, same structure as the QV GEMM.
// ---------------------------------------------------------------------------
__global__ void __launch_bounds__(256)
k_out_gemm(const __bf16* __restrict__ a, const __bf16* __restrict__ wT,
           const float* __restrict__ b_out, float* __restrict__ y) {
  int wave = blockIdx.x * (blockDim.x >> 5) + (threadIdx.x >> 5);
  int tm = wave & 127;        // ROWS/64 = 128 row tiles
  int tn = wave >> 7;         // DIM/64 = 7 col tiles
  int row0 = tm * 64, col0 = tn * 64;

  v8f acc[4][4] = {};
#pragma unroll 1
  for (int k = 0; k < DIM; k += 32) {
    v16bf fa[4], fb[4];
#pragma unroll
    for (int mi = 0; mi < 4; ++mi)
      fa[mi] = load_frag(a + (size_t)(row0 + mi * 16) * DIM + k, DIM);
#pragma unroll
    for (int ni = 0; ni < 4; ++ni)
      fb[ni] = load_frag(wT + (size_t)(col0 + ni * 16) * DIM + k, DIM);
#pragma unroll
    for (int mi = 0; mi < 4; ++mi)
#pragma unroll
      for (int ni = 0; ni < 4; ++ni)
        acc[mi][ni] = wmma_bf16(fa[mi], fb[ni], acc[mi][ni]);
  }

  int lane = threadIdx.x & 31;
  int c = lane & 15, hh = lane >> 4;
#pragma unroll
  for (int mi = 0; mi < 4; ++mi)
#pragma unroll
    for (int ni = 0; ni < 4; ++ni) {
      int cg = col0 + ni * 16 + c;
      float bv = b_out[cg];
#pragma unroll
      for (int j = 0; j < 8; ++j) {
        int rg = row0 + mi * 16 + j + 8 * hh;
        y[(size_t)rg * DIM + cg] = acc[mi][ni][j] + bv;
      }
    }
}

// ---------------------------------------------------------------------------
extern "C" void kernel_launch(void* const* d_in, const int* in_sizes, int n_in,
                              void* d_out, int out_size, void* d_ws, size_t ws_size,
                              hipStream_t stream) {
  const float* x        = (const float*)d_in[0];
  const float* w_qv     = (const float*)d_in[1];
  const float* ext_k    = (const float*)d_in[2];
  const float* ext_bias = (const float*)d_in[3];
  const float* w_out    = (const float*)d_in[4];
  const float* b_out    = (const float*)d_in[5];
  float* y = (float*)d_out;

  char* ws = (char*)d_ws;
  size_t off = 0;
  auto carve = [&](size_t bytes) {
    char* p = ws + off;
    off += (bytes + 255) & ~(size_t)255;
    return p;
  };
  __bf16* x_bf  = (__bf16*)carve((size_t)ROWS * DIM * 2);
  __bf16* wqvT  = (__bf16*)carve((size_t)QV_N * DIM * 2);
  __bf16* kx_bf = (__bf16*)carve((size_t)HEADS * SEQ * DIM_HEAD * 2);
  __bf16* woutT = (__bf16*)carve((size_t)DIM * DIM * 2);
  __bf16* q_ws  = (__bf16*)carve((size_t)BATCH * HEADS * SEQ * DIM_HEAD * 2);
  __bf16* vT_ws = (__bf16*)carve((size_t)BATCH * HEADS * SEQ * DIM_HEAD * 2);
  __bf16* attn  = (__bf16*)carve((size_t)ROWS * INNER * 2);
  (void)ws_size; (void)in_sizes; (void)n_in; (void)out_size;

  const int T = 256;
  k_cvt_copy<<<(ROWS * DIM + T - 1) / T, T, 0, stream>>>(x, x_bf, ROWS * DIM);
  k_cvt_transpose<<<(DIM * QV_N + T - 1) / T, T, 0, stream>>>(w_qv, wqvT, DIM, QV_N);
  k_cvt_copy<<<(HEADS * SEQ * DIM_HEAD + T - 1) / T, T, 0, stream>>>(
      ext_k, kx_bf, HEADS * SEQ * DIM_HEAD);
  k_cvt_transpose<<<(DIM * DIM + T - 1) / T, T, 0, stream>>>(w_out, woutT, DIM, DIM);

  // 128*14 = 1792 wave-tiles / 8 waves per block = 224 blocks (exact)
  k_qv_gemm<<<224, 256, 0, stream>>>(x_bf, wqvT, q_ws, vT_ws);
  // b*h*(n/64) = 8*7*16 = 896 blocks, 4 waves each
  k_attn<<<896, 128, 0, stream>>>(q_ws, kx_bf, vT_ws, ext_bias, attn);
  // 128*7 = 896 wave-tiles / 8 = 112 blocks (exact)
  k_out_gemm<<<112, 256, 0, stream>>>(attn, woutT, b_out, y);
}
